// QuaternionPhasorBlock_84825604096566
// MI455X (gfx1250) — matrix operand order repaired and checked by reading, not compiled, not run on hardware
//
#include <hip/hip_runtime.h>
#include <math.h>

// ---------------------------------------------------------------------------
// QuaternionPhasorBlock on MI455X (gfx1250, wave32, WMMA, async-to-LDS)
// D = 1024, B = 4, L = 4096
// ---------------------------------------------------------------------------

#define DIM   1024
#define BATCH 4
#define SEQ   4096
#define ROWS  (BATCH * SEQ)          // 16384
#define NQ    (DIM / 4)              // 256 quaternions per row
#define NQ_TOTAL ((size_t)ROWS * NQ)

typedef __attribute__((ext_vector_type(16))) __bf16 v16bf;
typedef __attribute__((ext_vector_type(8)))  __bf16 v8bf;
typedef __attribute__((ext_vector_type(4)))  __bf16 v4bf;
typedef __attribute__((ext_vector_type(8)))  float  v8f;
typedef __attribute__((ext_vector_type(4)))  int    v4i;

union FragBF { v16bf v; v8bf h[2]; };

// ---------------- CDNA5 async global->LDS (ASYNCcnt path) -------------------

#if __has_builtin(__builtin_amdgcn_global_load_async_to_lds_b128)
#define HAVE_ASYNC_LDS 1
#else
#define HAVE_ASYNC_LDS 0
#endif

typedef __attribute__((address_space(1))) v4i gv4i_t;   // global int4
typedef __attribute__((address_space(3))) v4i lv4i_t;   // LDS int4

__device__ __forceinline__ void copy_b128_to_lds(const __bf16* g, __bf16* l) {
#if HAVE_ASYNC_LDS
    __builtin_amdgcn_global_load_async_to_lds_b128((gv4i_t*)(g), (lv4i_t*)(l), 0, 0);
#else
    *(v8bf*)l = *(const v8bf*)g;           // sync fallback: global->VGPR->LDS
#endif
}

__device__ __forceinline__ void wait_async_all() {
#if HAVE_ASYNC_LDS
#if __has_builtin(__builtin_amdgcn_s_wait_asynccnt)
    __builtin_amdgcn_s_wait_asynccnt(0);
#else
    asm volatile("s_wait_asynccnt 0" ::: "memory");
#endif
#endif
}

// ------------------- fp32 -> bf16 hi/lo split (one pass) --------------------

__global__ __launch_bounds__(256)
void split_bf16_kernel(const float* __restrict__ src, __bf16* __restrict__ hi,
                       __bf16* __restrict__ lo, int n4)
{
    int i = blockIdx.x * 256 + threadIdx.x;
    if (i >= n4) return;
    float4 f = ((const float4*)src)[i];
    __bf16 h0 = (__bf16)f.x, h1 = (__bf16)f.y, h2 = (__bf16)f.z, h3 = (__bf16)f.w;
    v4bf H = {h0, h1, h2, h3};
    v4bf L = {(__bf16)(f.x - (float)h0), (__bf16)(f.y - (float)h1),
              (__bf16)(f.z - (float)h2), (__bf16)(f.w - (float)h3)};
    ((v4bf*)hi)[i] = H;
    ((v4bf*)lo)[i] = L;
}

// ----------------------------- GEMM ----------------------------------------
// C[M,N] = (Ahi+Alo)[M,K] @ (Whi+Wlo)[N,K]^T + bias[N] (+ residual)
// bf16x3 split on the bf16 WMMA pipe; double-buffered async-to-LDS staging.

#define BM 128
#define BN 64
#define BK 32
#define LDT 40   // padded LDS k-stride (bf16): 80B rows, conflict-free b128

__global__ __launch_bounds__(256)
void gemm_bf16x3(const __bf16* __restrict__ Ahi, const __bf16* __restrict__ Alo,
                 const __bf16* __restrict__ Whi, const __bf16* __restrict__ Wlo,
                 const float* __restrict__ bias, const float* __restrict__ residual,
                 float* __restrict__ C, int N, int K)
{
    __shared__ __attribute__((aligned(16))) __bf16 sAhi[2][BM * LDT];
    __shared__ __attribute__((aligned(16))) __bf16 sAlo[2][BM * LDT];
    __shared__ __attribute__((aligned(16))) __bf16 sBhi[2][BN * LDT];
    __shared__ __attribute__((aligned(16))) __bf16 sBlo[2][BN * LDT];

    const int tid  = threadIdx.x;
    const int lane = tid & 31;
    const int wave = tid >> 5;        // 0..7
    const int half = lane >> 4;       // 0/1
    const int ln16 = lane & 15;
    const int wm   = wave >> 1;       // 0..3 : 32-row slab
    const int wn   = wave & 1;        // 0..1 : 32-col slab

    const int blockM = blockIdx.y * BM;
    const int blockN = blockIdx.x * BN;

    // stage one K-tile (bf16 hi/lo planes) into LDS buffer `buf`
    auto stage = [&](int buf, int k0) {
#pragma unroll
        for (int i = 0; i < 2; ++i) {
            int idx = i * 256 + tid;          // 512 b128 chunks per A plane
            int row = idx >> 2;
            int c8  = (idx & 3) * 8;
            size_t go = (size_t)(blockM + row) * K + k0 + c8;
            copy_b128_to_lds(Ahi + go, &sAhi[buf][row * LDT + c8]);
            copy_b128_to_lds(Alo + go, &sAlo[buf][row * LDT + c8]);
        }
        {
            int idx = tid;                    // 256 b128 chunks per B plane
            int row = idx >> 2;
            int c8  = (idx & 3) * 8;
            size_t go = (size_t)(blockN + row) * K + k0 + c8;
            copy_b128_to_lds(Whi + go, &sBhi[buf][row * LDT + c8]);
            copy_b128_to_lds(Wlo + go, &sBlo[buf][row * LDT + c8]);
        }
    };

    stage(0, 0);
    wait_async_all();
    __syncthreads();

    v8f acc[2][2] = {};
    int cur = 0;
    for (int k0 = 0; k0 < K; k0 += BK) {
        const bool has_next = (k0 + BK) < K;
        if (has_next) stage(cur ^ 1, k0 + BK);   // async copies overlap WMMAs

        // ---- fragments (contiguous k order, k = 16*half + s) ------------------------
        FragBF ahi[2], alo[2], bhi[2], blo[2];
#pragma unroll
        for (int mt = 0; mt < 2; ++mt) {
            int row = wm * 32 + mt * 16 + ln16;
            int ka  = half * 16;  // lanes0-15: K0..15; lanes16-31: K16..31
            ahi[mt].h[0] = *(const v8bf*)&sAhi[cur][row * LDT + ka];
            ahi[mt].h[1] = *(const v8bf*)&sAhi[cur][row * LDT + ka + 8];
            alo[mt].h[0] = *(const v8bf*)&sAlo[cur][row * LDT + ka];
            alo[mt].h[1] = *(const v8bf*)&sAlo[cur][row * LDT + ka + 8];
        }
#pragma unroll
        for (int nt = 0; nt < 2; ++nt) {
            int col = wn * 32 + nt * 16 + ln16;
            int kb  = half * 16;  // lanes0-15: K0..15; lanes16-31: K16..31
            bhi[nt].h[0] = *(const v8bf*)&sBhi[cur][col * LDT + kb];
            bhi[nt].h[1] = *(const v8bf*)&sBhi[cur][col * LDT + kb + 8];
            blo[nt].h[0] = *(const v8bf*)&sBlo[cur][col * LDT + kb];
            blo[nt].h[1] = *(const v8bf*)&sBlo[cur][col * LDT + kb + 8];
        }

        // ---- 12 WMMAs per K-step (bf16x3) ---------------------------------
#pragma unroll
        for (int mt = 0; mt < 2; ++mt)
#pragma unroll
            for (int nt = 0; nt < 2; ++nt) {
                acc[mt][nt] = __builtin_amdgcn_wmma_f32_16x16x32_bf16(
                    false, alo[mt].v, false, bhi[nt].v, (short)0, acc[mt][nt], false, false);
                acc[mt][nt] = __builtin_amdgcn_wmma_f32_16x16x32_bf16(
                    false, ahi[mt].v, false, blo[nt].v, (short)0, acc[mt][nt], false, false);
                acc[mt][nt] = __builtin_amdgcn_wmma_f32_16x16x32_bf16(
                    false, ahi[mt].v, false, bhi[nt].v, (short)0, acc[mt][nt], false, false);
            }

        if (has_next) wait_async_all();
        __syncthreads();
        cur ^= 1;
    }

    // ---- epilogue: D layout (m = r + 8*half, n = lane%16) ------------------
#pragma unroll
    for (int mt = 0; mt < 2; ++mt)
#pragma unroll
        for (int nt = 0; nt < 2; ++nt) {
            int n = blockN + wn * 32 + nt * 16 + ln16;
            float bv = bias[n];
#pragma unroll
            for (int r = 0; r < 8; ++r) {
                int m = blockM + wm * 32 + mt * 16 + r + 8 * half;
                float v = acc[mt][nt][r] + bv;
                if (residual) v += residual[(size_t)m * N + n];
                C[(size_t)m * N + n] = v;
            }
        }
}

// ------------------- keys normalize + bind = values (x) keys ----------------

__device__ __forceinline__ float4 qmul(float4 a, float4 b) {
    return make_float4(
        a.x * b.x - a.y * b.y - a.z * b.z - a.w * b.w,
        a.x * b.y + a.y * b.x + a.z * b.w - a.w * b.z,
        a.x * b.z - a.y * b.w + a.z * b.x + a.w * b.y,
        a.x * b.w + a.y * b.z - a.z * b.y + a.w * b.x);
}

__global__ __launch_bounds__(256)
void bind_kernel(float* __restrict__ keys, float* __restrict__ vals_bound)
{
    size_t i = (size_t)blockIdx.x * 256 + threadIdx.x;   // quaternion index
    if (i >= NQ_TOTAL) return;
    float4 k = *(const float4*)(keys + i * 4);
    float nrm = sqrtf(k.x * k.x + k.y * k.y + k.z * k.z + k.w * k.w);
    float s = 1.0f / fmaxf(nrm, 1e-12f);
    k.x *= s; k.y *= s; k.z *= s; k.w *= s;
    *(float4*)(keys + i * 4) = k;                        // keep normalized keys
    float4 v = *(const float4*)(vals_bound + i * 4);
    *(float4*)(vals_bound + i * 4) = qmul(v, k);         // bound, in place
}

// ------------------------- cumsum along L (in place) ------------------------
// one block per (b, q) quaternion channel; float4 Hillis-Steele with carry.

__global__ __launch_bounds__(256)
void cumsum4_kernel(float* __restrict__ data)
{
    const int b = blockIdx.x / NQ;
    const int q = blockIdx.x % NQ;
    float4* base = (float4*)data + (size_t)b * SEQ * NQ + q;
    __shared__ float4 s[256];
    const int tid = threadIdx.x;
    float4 carry = make_float4(0.f, 0.f, 0.f, 0.f);
    for (int chunk = 0; chunk < SEQ; chunk += 256) {
        float4 v = base[(size_t)(chunk + tid) * NQ];
        s[tid] = v;
        __syncthreads();
#pragma unroll
        for (int off = 1; off < 256; off <<= 1) {
            float4 t = (tid >= off) ? s[tid - off] : make_float4(0.f, 0.f, 0.f, 0.f);
            __syncthreads();
            float4 a = s[tid];
            a.x += t.x; a.y += t.y; a.z += t.z; a.w += t.w;
            s[tid] = a;
            __syncthreads();
        }
        float4 total = s[255];
        float4 o = s[tid];
        o.x += carry.x; o.y += carry.y; o.z += carry.z; o.w += carry.w;
        base[(size_t)(chunk + tid) * NQ] = o;
        carry.x += total.x; carry.y += total.y; carry.z += total.z; carry.w += total.w;
        __syncthreads();
    }
}

// -------- retrieved = memory (x) conj(keys), /sqrt(pos), LayerNorm ----------

__global__ __launch_bounds__(256)
void retrieve_ln_kernel(const float* __restrict__ memory, const float* __restrict__ keys,
                        const float* __restrict__ ln_g, const float* __restrict__ ln_b,
                        float* __restrict__ normed)
{
    const int bl = blockIdx.x;                 // row index in [0, B*L)
    const int l  = bl % SEQ;
    const int q  = threadIdx.x;                // quaternion in row
    const size_t rowoff = (size_t)bl * DIM;

    float4 m = *(const float4*)(memory + rowoff + q * 4);
    float4 k = *(const float4*)(keys   + rowoff + q * 4);
    float4 kc = make_float4(k.x, -k.y, -k.z, -k.w);
    float4 y = qmul(m, kc);
    float ps = rsqrtf((float)(l + 1));
    y.x *= ps; y.y *= ps; y.z *= ps; y.w *= ps;

    __shared__ float rs[256], rq[256];
    rs[q] = y.x + y.y + y.z + y.w;
    rq[q] = y.x * y.x + y.y * y.y + y.z * y.z + y.w * y.w;
    __syncthreads();
#pragma unroll
    for (int off = 128; off > 0; off >>= 1) {
        if (q < off) { rs[q] += rs[q + off]; rq[q] += rq[q + off]; }
        __syncthreads();
    }
    float mu  = rs[0] * (1.0f / DIM);
    float var = rq[0] * (1.0f / DIM) - mu * mu;
    float inv = rsqrtf(var + 1e-5f);

    float4 g  = *(const float4*)(ln_g + q * 4);
    float4 bb = *(const float4*)(ln_b + q * 4);
    float4 o;
    o.x = (y.x - mu) * inv * g.x + bb.x;
    o.y = (y.y - mu) * inv * g.y + bb.y;
    o.z = (y.z - mu) * inv * g.z + bb.z;
    o.w = (y.w - mu) * inv * g.w + bb.w;
    *(float4*)(normed + rowoff + q * 4) = o;
}

// ---------------------------------------------------------------------------

extern "C" void kernel_launch(void* const* d_in, const int* in_sizes, int n_in,
                              void* d_out, int out_size, void* d_ws, size_t ws_size,
                              hipStream_t stream) {
    (void)in_sizes; (void)n_in; (void)out_size; (void)ws_size;
    const float* x    = (const float*)d_in[0];
    const float* Wk   = (const float*)d_in[1];
    const float* bk   = (const float*)d_in[2];
    const float* Wv   = (const float*)d_in[3];
    const float* bv   = (const float*)d_in[4];
    const float* ln_g = (const float*)d_in[5];
    const float* ln_b = (const float*)d_in[6];
    const float* Wo   = (const float*)d_in[7];
    const float* bo   = (const float*)d_in[8];
    float* out = (float*)d_out;

    const size_t ELEMS = (size_t)ROWS * DIM;   // 16,777,216
    const size_t WEL   = (size_t)DIM * DIM;    // 1,048,576

    char* p = (char*)d_ws;
    float*  wsK  = (float*)p;  p += ELEMS * sizeof(float);   // keys (normalized)
    float*  wsV  = (float*)p;  p += ELEMS * sizeof(float);   // values->bound->memory
    float*  wsN  = (float*)p;  p += ELEMS * sizeof(float);   // normed
    __bf16* xhi  = (__bf16*)p; p += ELEMS * sizeof(__bf16);
    __bf16* xlo  = (__bf16*)p; p += ELEMS * sizeof(__bf16);
    __bf16* nhi  = (__bf16*)p; p += ELEMS * sizeof(__bf16);
    __bf16* nlo  = (__bf16*)p; p += ELEMS * sizeof(__bf16);
    __bf16* wkhi = (__bf16*)p; p += WEL * sizeof(__bf16);
    __bf16* wklo = (__bf16*)p; p += WEL * sizeof(__bf16);
    __bf16* wvhi = (__bf16*)p; p += WEL * sizeof(__bf16);
    __bf16* wvlo = (__bf16*)p; p += WEL * sizeof(__bf16);
    __bf16* wohi = (__bf16*)p; p += WEL * sizeof(__bf16);
    __bf16* wolo = (__bf16*)p; p += WEL * sizeof(__bf16);

    // one-pass fp32 -> bf16 hi/lo splits
    split_bf16_kernel<<<(int)(ELEMS / 4 / 256), 256, 0, stream>>>(x,  xhi,  xlo,  (int)(ELEMS / 4));
    split_bf16_kernel<<<(int)(WEL / 4 / 256),   256, 0, stream>>>(Wk, wkhi, wklo, (int)(WEL / 4));
    split_bf16_kernel<<<(int)(WEL / 4 / 256),   256, 0, stream>>>(Wv, wvhi, wvlo, (int)(WEL / 4));
    split_bf16_kernel<<<(int)(WEL / 4 / 256),   256, 0, stream>>>(Wo, wohi, wolo, (int)(WEL / 4));

    dim3 gemm_grid(DIM / BN, ROWS / BM);       // (16, 128)

    // keys = x @ Wk^T + bk ; values = x @ Wv^T + bv
    gemm_bf16x3<<<gemm_grid, 256, 0, stream>>>(xhi, xlo, wkhi, wklo, bk, nullptr, wsK, DIM, DIM);
    gemm_bf16x3<<<gemm_grid, 256, 0, stream>>>(xhi, xlo, wvhi, wvlo, bv, nullptr, wsV, DIM, DIM);

    // normalize keys, bound = qmul(values, keys) (in place over values)
    bind_kernel<<<(int)((NQ_TOTAL + 255) / 256), 256, 0, stream>>>(wsK, wsV);

    // memory = cumsum_L(bound), in place (L2-resident)
    cumsum4_kernel<<<BATCH * NQ, 256, 0, stream>>>(wsV);

    // retrieved = qmul(memory, conj(keys)) / sqrt(pos); LayerNorm -> wsN
    retrieve_ln_kernel<<<ROWS, 256, 0, stream>>>(wsV, wsK, ln_g, ln_b, wsN);

    // out = x + normed @ Wo^T + bo
    split_bf16_kernel<<<(int)(ELEMS / 4 / 256), 256, 0, stream>>>(wsN, nhi, nlo, (int)(ELEMS / 4));
    gemm_bf16x3<<<gemm_grid, 256, 0, stream>>>(nhi, nlo, wohi, wolo, bo, x, out, DIM, DIM);
}
